// GlobalMixtureEncoder_67551245631803
// MI455X (gfx1250) — compile-verified, matrix operand order checked by reading, not run on hardware
//
#include <hip/hip_runtime.h>
#include <cstdint>
#include <cstddef>

typedef __bf16 bf16;
typedef __attribute__((ext_vector_type(16))) __bf16 v16bf;
typedef __attribute__((ext_vector_type(8)))  float  v8f;
typedef __attribute__((ext_vector_type(4)))  unsigned int u32x4;

typedef __attribute__((address_space(3))) bf16 lds_bf16;

#define BETA_F 0.001f

// ---------------------------------------------------------------------------
// CDNA5 async global->LDS copy (ASYNCcnt path) via inline asm.
// ISA: GLOBAL_LOAD_ASYNC_TO_LDS_B128 — vdst VGPR holds the LDS byte address,
// vaddr pair holds the 64-bit global address (GV mode, saddr = off).
// ---------------------------------------------------------------------------
__device__ inline void async_ld16(bf16* lds, const bf16* g) {
  asm volatile("global_load_async_to_lds_b128 %0, %1, off"
               :
               : "v"((lds_bf16*)lds), "v"(g)
               : "memory");
}

__device__ inline void wait_async(int pending8) {
#if __has_builtin(__builtin_amdgcn_s_wait_asynccnt)
  if (pending8) __builtin_amdgcn_s_wait_asynccnt(8);
  else          __builtin_amdgcn_s_wait_asynccnt(0);
#else
  if (pending8) asm volatile("s_wait_asynccnt 8" ::: "memory");
  else          asm volatile("s_wait_asynccnt 0" ::: "memory");
#endif
}

// ---------------------------------------------------------------------------
// WMMA wrapper: D = A(bf16 16x32) * B(bf16 32x16) + C(f32 16x16)
// ---------------------------------------------------------------------------
__device__ inline v8f wmma_bf16f32(v16bf a, v16bf b, v8f c) {
  return __builtin_amdgcn_wmma_f32_16x16x32_bf16(
      /*neg_a=*/false, a, /*neg_b=*/false, b,
      /*c_mod=*/(short)0, c, /*reuse_a=*/false, /*reuse_b=*/false);
}

// Load one 16x32 bf16 A-fragment (per-lane) from a 128x32 row-major LDS tile.
// Lane layout (ISA 7.12.2, 16-bit A 16x32):
//   lanes 0-15:  M=lane,    VGPR0..3 = K 0..7,   VGPR4..7 = K 16..23
//   lanes 16-31: M=lane-16, VGPR0..3 = K 8..15,  VGPR4..7 = K 24..31
__device__ inline v16bf load_frag(const bf16* __restrict__ s, int row, int half) {
  union { u32x4 u[2]; v16bf v; } f;
  const bf16* p = s + row * 32 + half * 8;
  f.u[0] = *(const u32x4*)(p);        // K = 8*half .. 8*half+7
  f.u[1] = *(const u32x4*)(p + 16);   // K = 16+8*half .. +7
  return f.v;
}

// ---------------------------------------------------------------------------
// GEMM: C(MxN) = A(MxKd) * B^T  with A,B given as bf16 hi/lo pairs, both
// stored K-contiguous (A row-major MxKd; B stored as NxKd row-major).
// Hi/lo split: acc += aL*bH + aH*bL + aH*bH  (~fp32 accuracy).
// EPI 0: store f32 C.  EPI 1: split-store bf16 hi/lo.  EPI 2: fused MSE loss.
// Block = 256 threads (8 waves), tile 128x128; wave grid 4(M) x 2(N),
// wave tile 32x64 = 2x4 16x16 fragments. Double-buffered async LDS staging.
// ---------------------------------------------------------------------------
template <int EPI>
__global__ __launch_bounds__(256) void gemm_hilo(
    const bf16* __restrict__ Ahi, const bf16* __restrict__ Alo,
    const bf16* __restrict__ Bhi, const bf16* __restrict__ Blo,
    int M, int N, int Kd,
    float* __restrict__ Cf,
    bf16* __restrict__ Chi, bf16* __restrict__ Clo,
    const float* __restrict__ ref, float* __restrict__ outv, float invN) {
  __shared__ __align__(16) bf16 sAh[2][128 * 32];
  __shared__ __align__(16) bf16 sAl[2][128 * 32];
  __shared__ __align__(16) bf16 sBh[2][128 * 32];
  __shared__ __align__(16) bf16 sBl[2][128 * 32];
  __shared__ float lossRow[128];

  const int tid  = threadIdx.x;
  const int lane = tid & 31;
  const int wave = tid >> 5;
  const int wM   = wave & 3;   // 0..3
  const int wN   = wave >> 2;  // 0..1
  const int m    = lane & 15;
  const int hf   = lane >> 4;
  const int blockM = blockIdx.y * 128;
  const int blockN = blockIdx.x * 128;

  v8f acc[2][4];
#pragma unroll
  for (int f = 0; f < 2; ++f)
#pragma unroll
    for (int g = 0; g < 4; ++g) acc[f][g] = (v8f){0.f,0.f,0.f,0.f,0.f,0.f,0.f,0.f};

  // Tile staging: each 128x32 bf16 tile = 8KB = 512 x 16B chunks.
  // 256 threads * 2 chunks each: row = h*64 + (tid>>2), 16B quarter q = tid&3.
  const int qrow = tid >> 2;  // 0..63
  const int q    = tid & 3;   // 0..3

  auto compute = [&](int buf) {
    v16bf aH[2], aL[2], bH[4], bL[4];
#pragma unroll
    for (int f = 0; f < 2; ++f) {
      int r = wM * 32 + f * 16 + m;
      aH[f] = load_frag(sAh[buf], r, hf);
      aL[f] = load_frag(sAl[buf], r, hf);
    }
#pragma unroll
    for (int g = 0; g < 4; ++g) {
      int r = wN * 64 + g * 16 + m;
      bH[g] = load_frag(sBh[buf], r, hf);
      bL[g] = load_frag(sBl[buf], r, hf);
    }
#pragma unroll
    for (int f = 0; f < 2; ++f)
#pragma unroll
      for (int g = 0; g < 4; ++g) {
        acc[f][g] = wmma_bf16f32(aL[f], bH[g], acc[f][g]);
        acc[f][g] = wmma_bf16f32(aH[f], bL[g], acc[f][g]);
        acc[f][g] = wmma_bf16f32(aH[f], bH[g], acc[f][g]);
      }
  };

  // Async double-buffered pipeline: 8 async b128 copies per thread per stage.
  auto issue = [&](int kk, int buf) {
#pragma unroll
    for (int h = 0; h < 2; ++h) {
      int row = h * 64 + qrow;
      int so  = row * 32 + q * 8;
      size_t aoff = (size_t)(blockM + row) * Kd + kk + q * 8;
      size_t boff = (size_t)(blockN + row) * Kd + kk + q * 8;
      async_ld16(&sAh[buf][so], Ahi + aoff);
      async_ld16(&sAl[buf][so], Alo + aoff);
      async_ld16(&sBh[buf][so], Bhi + boff);
      async_ld16(&sBl[buf][so], Blo + boff);
    }
  };
  const int nsteps = Kd / 32;
  issue(0, 0);
  for (int it = 0; it < nsteps; ++it) {
    const int cur = it & 1;
    const bool more = (it + 1) < nsteps;
    if (more) issue((it + 1) * 32, cur ^ 1);
    // Async loads complete in order: <=8 outstanding means stage `cur` landed.
    wait_async(more ? 1 : 0);
    __syncthreads();
    compute(cur);
    __syncthreads();  // all waves done reading buf `cur` before it is reused
  }

  if (EPI == 2) {
    if (tid < 128) lossRow[tid] = 0.0f;
    __syncthreads();
  }

  // C fragment layout (ISA 7.12.2): VGPR r, lanes0-15 -> M=r,  N=lane;
  // lanes16-31 -> M=8+r, N=lane-16.
#pragma unroll
  for (int f = 0; f < 2; ++f)
#pragma unroll
    for (int g = 0; g < 4; ++g) {
      int colL = wN * 64 + g * 16 + m;
#pragma unroll
      for (int r = 0; r < 8; ++r) {
        int rowL   = wM * 32 + f * 16 + hf * 8 + r;
        size_t idx = (size_t)(blockM + rowL) * N + (blockN + colL);
        float x    = acc[f][g][r];
        if (EPI == 0) {
          Cf[idx] = x;
        } else if (EPI == 1) {
          bf16 h   = (bf16)x;
          Chi[idx] = h;
          Clo[idx] = (bf16)(x - (float)h);
        } else {
          float d = x - ref[idx];
          atomicAdd(&lossRow[rowL], d * d);
        }
      }
    }

  if (EPI == 2) {
    __syncthreads();
    if (tid < 128) atomicAdd(&outv[blockM + tid], lossRow[tid] * invN);
  }
}

// ---------------------------------------------------------------------------
// Row softmax over Kc=2048 columns: p = softmax(S*scale - bias), then split
// p into bf16 hi/lo for the following WMMA GEMM. One block per row.
// (The per-row x2/lat2 terms of _neg_mean_sqdist cancel inside softmax.)
// ---------------------------------------------------------------------------
__global__ __launch_bounds__(256) void softmax_rows(
    const float* __restrict__ S, const float* __restrict__ bias, float scale,
    bf16* __restrict__ Phi, bf16* __restrict__ Plo, int Kc) {
  __shared__ float red[256];
  const int t     = threadIdx.x;
  const size_t base = (size_t)blockIdx.x * Kc;
  const int per   = 8;  // Kc == 2048, 256 threads
  float v[8];
  float mx = -3.402823e38f;
#pragma unroll
  for (int i = 0; i < per; ++i) {
    int k = t + i * 256;
    float x = S[base + k] * scale - bias[k];
    v[i] = x;
    mx = fmaxf(mx, x);
  }
  red[t] = mx;
  __syncthreads();
  for (int s = 128; s > 0; s >>= 1) {
    if (t < s) red[t] = fmaxf(red[t], red[t + s]);
    __syncthreads();
  }
  mx = red[0];
  __syncthreads();
  float sum = 0.f;
#pragma unroll
  for (int i = 0; i < per; ++i) {
    v[i] = __expf(v[i] - mx);
    sum += v[i];
  }
  red[t] = sum;
  __syncthreads();
  for (int s = 128; s > 0; s >>= 1) {
    if (t < s) red[t] += red[t + s];
    __syncthreads();
  }
  float inv = 1.0f / red[0];
#pragma unroll
  for (int i = 0; i < per; ++i) {
    int k = t + i * 256;
    float p = v[i] * inv;
    bf16 h = (bf16)p;
    Phi[base + k] = h;
    Plo[base + k] = (bf16)(p - (float)h);
  }
}

// ---------------------------------------------------------------------------
// Prep kernels
// ---------------------------------------------------------------------------
__global__ void split_bf16(const float* __restrict__ in, bf16* __restrict__ hi,
                           bf16* __restrict__ lo, size_t n) {
  size_t i = (size_t)blockIdx.x * blockDim.x + threadIdx.x;
  size_t stride = (size_t)gridDim.x * blockDim.x;
  for (; i < n; i += stride) {
    float x = in[i];
    bf16 h = (bf16)x;
    hi[i] = h;
    lo[i] = (bf16)(x - (float)h);
  }
}

// out(CxR) = in(RxC)^T, split to bf16 hi/lo. 32x32 LDS tiles.
__global__ void transpose_split(const float* __restrict__ in,
                                bf16* __restrict__ hi, bf16* __restrict__ lo,
                                int R, int C) {
  __shared__ float tile[32][33];
  const int c0 = blockIdx.x * 32, r0 = blockIdx.y * 32;
  const int tx = threadIdx.x;
  for (int j = threadIdx.y; j < 32; j += 8)
    tile[j][tx] = in[(size_t)(r0 + j) * C + c0 + tx];
  __syncthreads();
  for (int j = threadIdx.y; j < 32; j += 8) {
    float x = tile[tx][j];  // = in[r0+tx][c0+j]
    size_t o = (size_t)(c0 + j) * R + r0 + tx;
    bf16 h = (bf16)x;
    hi[o] = h;
    lo[o] = (bf16)(x - (float)h);
  }
}

// bias[k] = BETA * mean_d(in[k,:]^2), one block per row k.
__global__ void rowsq_mean(const float* __restrict__ in, float* __restrict__ outb,
                           int D) {
  __shared__ float red[256];
  const int k = blockIdx.x, t = threadIdx.x;
  const float* row = in + (size_t)k * D;
  float s = 0.f;
  for (int i = t; i < D; i += 256) { float x = row[i]; s += x * x; }
  red[t] = s;
  __syncthreads();
  for (int st = 128; st > 0; st >>= 1) {
    if (t < st) red[t] += red[t + st];
    __syncthreads();
  }
  if (t == 0) outb[k] = BETA_F * red[0] / (float)D;
}

// bias[k] = BETA * mean_e(in[:,k]^2) for in (ExK) row-major (coalesced over k).
__global__ void colsq_mean(const float* __restrict__ in, float* __restrict__ outb,
                           int E, int K) {
  int k = blockIdx.x * blockDim.x + threadIdx.x;
  if (k >= K) return;
  float s = 0.f;
  for (int e = 0; e < E; ++e) { float x = in[(size_t)e * K + k]; s += x * x; }
  outb[k] = BETA_F * s / (float)E;
}

__global__ void zero_f32(float* __restrict__ p, int n) {
  int i = blockIdx.x * blockDim.x + threadIdx.x;
  if (i < n) p[i] = 0.0f;
}

// ---------------------------------------------------------------------------
// Launch
// ---------------------------------------------------------------------------
extern "C" void kernel_launch(void* const* d_in, const int* in_sizes, int n_in,
                              void* d_out, int out_size, void* d_ws,
                              size_t ws_size, hipStream_t stream) {
  (void)in_sizes; (void)n_in; (void)out_size; (void)ws_size;
  const float* images    = (const float*)d_in[0];  // (B, D)
  const float* project_w = (const float*)d_in[1];  // (K, D)
  const float* rec_w     = (const float*)d_in[2];  // (E, K)
  float* out = (float*)d_out;                      // (B,)

  const int Bn = 8192, Dn = 1024, Kn = 2048, En = 512;

  char* ws = (char*)d_ws;
  size_t off = 0;
  auto carve = [&](size_t bytes) -> char* {
    char* p = ws + off;
    off += (bytes + 255) & ~(size_t)255;
    return p;
  };
  bf16*  img_hi = (bf16*)carve((size_t)Bn * Dn * 2);
  bf16*  img_lo = (bf16*)carve((size_t)Bn * Dn * 2);
  bf16*  pw_hi  = (bf16*)carve((size_t)Kn * Dn * 2);   // project_w rows (K x D)
  bf16*  pw_lo  = (bf16*)carve((size_t)Kn * Dn * 2);
  bf16*  pwT_hi = (bf16*)carve((size_t)Dn * Kn * 2);   // project_w^T (D x K)
  bf16*  pwT_lo = (bf16*)carve((size_t)Dn * Kn * 2);
  bf16*  rw_hi  = (bf16*)carve((size_t)En * Kn * 2);   // rec_w rows (E x K)
  bf16*  rw_lo  = (bf16*)carve((size_t)En * Kn * 2);
  bf16*  rwT_hi = (bf16*)carve((size_t)Kn * En * 2);   // rec_w^T (K x E)
  bf16*  rwT_lo = (bf16*)carve((size_t)Kn * En * 2);
  float* S      = (float*)carve((size_t)Bn * Kn * 4);  // logits scratch
  bf16*  p_hi   = (bf16*)carve((size_t)Bn * Kn * 2);   // xp then yp
  bf16*  p_lo   = (bf16*)carve((size_t)Bn * Kn * 2);
  bf16*  lat_hi = (bf16*)carve((size_t)Bn * En * 2);
  bf16*  lat_lo = (bf16*)carve((size_t)Bn * En * 2);
  float* biasC  = (float*)carve((size_t)Kn * 4);       // BETA * c2[k]
  float* biasR  = (float*)carve((size_t)Kn * 4);       // BETA * r2[k]

  // --- prep ---
  split_bf16<<<2048, 256, 0, stream>>>(images,    img_hi, img_lo, (size_t)Bn * Dn);
  split_bf16<<<1024, 256, 0, stream>>>(project_w, pw_hi,  pw_lo,  (size_t)Kn * Dn);
  split_bf16<<<512,  256, 0, stream>>>(rec_w,     rw_hi,  rw_lo,  (size_t)En * Kn);
  transpose_split<<<dim3(Dn / 32, Kn / 32), dim3(32, 8), 0, stream>>>(
      project_w, pwT_hi, pwT_lo, Kn, Dn);
  transpose_split<<<dim3(Kn / 32, En / 32), dim3(32, 8), 0, stream>>>(
      rec_w, rwT_hi, rwT_lo, En, Kn);
  rowsq_mean<<<Kn, 256, 0, stream>>>(project_w, biasC, Dn);
  colsq_mean<<<Kn / 256, 256, 0, stream>>>(rec_w, biasR, En, Kn);

  // --- encode: S1 = images @ project_w^T ; xp = softmax(BETA*(2/D*S1 - c2)) ---
  gemm_hilo<0><<<dim3(Kn / 128, Bn / 128), 256, 0, stream>>>(
      img_hi, img_lo, pw_hi, pw_lo, Bn, Kn, Dn,
      S, nullptr, nullptr, nullptr, nullptr, 0.f);
  softmax_rows<<<Bn, 256, 0, stream>>>(S, biasC, BETA_F * 2.0f / (float)Dn,
                                       p_hi, p_lo, Kn);

  // --- lat = xp @ rec_w^T (split-stored bf16) ---
  gemm_hilo<1><<<dim3(En / 128, Bn / 128), 256, 0, stream>>>(
      p_hi, p_lo, rw_hi, rw_lo, Bn, En, Kn,
      nullptr, lat_hi, lat_lo, nullptr, nullptr, 0.f);

  // --- decode: S2 = lat @ rec_w ; yp = softmax(BETA*(2/E*S2 - r2)) ---
  gemm_hilo<0><<<dim3(Kn / 128, Bn / 128), 256, 0, stream>>>(
      lat_hi, lat_lo, rwT_hi, rwT_lo, Bn, Kn, En,
      S, nullptr, nullptr, nullptr, nullptr, 0.f);
  softmax_rows<<<Bn, 256, 0, stream>>>(S, biasR, BETA_F * 2.0f / (float)En,
                                       p_hi, p_lo, Kn);

  // --- recon = yp @ project_w, fused per-row MSE vs images -> out ---
  zero_f32<<<Bn / 256, 256, 0, stream>>>(out, Bn);
  gemm_hilo<2><<<dim3(Dn / 128, Bn / 128), 256, 0, stream>>>(
      p_hi, p_lo, pwT_hi, pwT_lo, Bn, Dn, Kn,
      nullptr, nullptr, nullptr, images, out, 1.0f / (float)Dn);
}
